// PAST_77283641524379
// MI455X (gfx1250) — matrix operand order repaired and checked by reading, not compiled, not run on hardware
//
#include <hip/hip_runtime.h>
#include <stdint.h>

#define N_SPOTS 8192
#define D_IN    3000
#define D_LAT   50
#define D_PAD   64            // latent dim padded to 4x16 WMMA tiles
#define KCH     32            // bf16 WMMA K per step
#define ENC_KCH 128           // K per staged chunk (4 WMMA steps)
#define ENC_KCHUNKS 24        // 3072 / 128
#define ENC_KPAD 3072         // padded K for encoder (x and WencT)
#define DEC_OSTRIDE 3072      // 24 blocks * 128 cols
#define MAX_NBR 16

typedef __bf16 bf16_t;
typedef bf16_t v16bf __attribute__((ext_vector_type(16)));
typedef float  v8f   __attribute__((ext_vector_type(8)));

// ---------------------------------------------------------------------------
// CDNA5 async global->LDS copies (ASYNCcnt) with sync fallback.
// Probe result: builtin takes (int4 AS1*, int4 AS3*, imm offset, imm cpol).
// ---------------------------------------------------------------------------
#if defined(__has_builtin)
#if __has_builtin(__builtin_amdgcn_global_load_async_to_lds_b128) && \
    __has_builtin(__builtin_amdgcn_s_wait_asynccnt)
#define HAVE_ASYNC_LDS 1
#endif
#endif
#ifndef HAVE_ASYNC_LDS
#define HAVE_ASYNC_LDS 0
#endif

typedef int v4i_t __attribute__((vector_size(16)));
typedef v4i_t __attribute__((address_space(1)))* gbl_v4i_p;
typedef v4i_t __attribute__((address_space(3)))* lds_v4i_p;

__device__ __forceinline__ void async_copy16(const void* g, void* l) {
#if HAVE_ASYNC_LDS
  __builtin_amdgcn_global_load_async_to_lds_b128((gbl_v4i_p)g, (lds_v4i_p)l, 0, 0);
#else
  *(uint4*)l = *(const uint4*)g;
#endif
}
__device__ __forceinline__ void wait_async_lds() {
#if HAVE_ASYNC_LDS
  __builtin_amdgcn_s_wait_asynccnt(0);
#endif
}

// ---------------------------------------------------------------------------
// Weight prep: pack [enc_fc1;enc_fc2] -> WencT[3072][64] bf16 (K-major),
// bias pad [64], dec_fc_w -> WdecT[64][3072] bf16 (K-major). Pads zeroed.
// ---------------------------------------------------------------------------
__global__ void prep_weights(const float* __restrict__ fc1w, const float* __restrict__ fc1b,
                             const float* __restrict__ fc2w, const float* __restrict__ decw,
                             bf16_t* __restrict__ wencT, float* __restrict__ biasPad,
                             bf16_t* __restrict__ wdecT) {
  const int enc_total = ENC_KPAD * D_PAD;
  const int total = enc_total + D_PAD + D_PAD * DEC_OSTRIDE;
  for (int idx = blockIdx.x * blockDim.x + threadIdx.x; idx < total;
       idx += gridDim.x * blockDim.x) {
    if (idx < enc_total) {
      int k = idx >> 6;
      int n = idx & 63;
      float v = 0.f;
      if (k < D_IN && n < D_LAT)
        v = (n < 10) ? fc1w[n * D_IN + k] : fc2w[(n - 10) * D_IN + k];
      wencT[idx] = (bf16_t)v;
    } else if (idx < enc_total + D_PAD) {
      int n = idx - enc_total;
      biasPad[n] = (n < 10) ? fc1b[n] : 0.f;
    } else {
      int r = idx - (enc_total + D_PAD);
      int k = r / DEC_OSTRIDE;
      int o = r % DEC_OSTRIDE;
      float v = (k < D_LAT && o < D_IN) ? decw[o * D_LAT + k] : 0.f;
      wdecT[(size_t)k * DEC_OSTRIDE + o] = (bf16_t)v;
    }
  }
}

// ---------------------------------------------------------------------------
// f32 -> bf16 convert with column padding. src_cols must be a multiple of 4.
// Used for x (3000 -> 3072-padded) and the decoder input (64 -> 64).
// ---------------------------------------------------------------------------
__global__ void convert_pad_bf16(const float* __restrict__ src, bf16_t* __restrict__ dst,
                                 int rows, int src_cols, int dst_stride) {
  const int qcols = dst_stride >> 2;
  const size_t total = (size_t)rows * qcols;
  for (size_t i = (size_t)blockIdx.x * blockDim.x + threadIdx.x; i < total;
       i += (size_t)gridDim.x * blockDim.x) {
    int r = (int)(i / qcols);
    int c4 = (int)(i % qcols) << 2;
    float4 v = make_float4(0.f, 0.f, 0.f, 0.f);
    if (c4 < src_cols) v = *(const float4*)(src + (size_t)r * src_cols + c4);
    bf16_t* d = dst + (size_t)r * dst_stride + c4;
    d[0] = (bf16_t)v.x; d[1] = (bf16_t)v.y; d[2] = (bf16_t)v.z; d[3] = (bf16_t)v.w;
  }
}

// ---------------------------------------------------------------------------
// Neighbor extraction: one wave32 per row, float4 loads (128 cols/iter),
// deterministic order-preserving compaction via shfl prefix-scan.
// ---------------------------------------------------------------------------
__global__ __launch_bounds__(256) void build_nbrs(const float* __restrict__ g,
                                                  int* __restrict__ nbr_idx,
                                                  int* __restrict__ nbr_cnt) {
  int wave = threadIdx.x >> 5;
  int lane = threadIdx.x & 31;
  int row = blockIdx.x * (blockDim.x >> 5) + wave;
  if (row >= N_SPOTS) return;
  const float* grow = g + (size_t)row * N_SPOTS;
  int cnt = 0;
  for (int c0 = 0; c0 < N_SPOTS; c0 += 128) {
    __builtin_prefetch(grow + c0 + 1024, 0, 0);
    const float4 v = *(const float4*)(grow + c0 + 4 * lane);
    int m4 = (v.x != 0.f) | ((v.y != 0.f) << 1) | ((v.z != 0.f) << 2) |
             ((v.w != 0.f) << 3);
    int tot = __popc(m4);
    // exclusive prefix across lanes
    int pre = tot;
#pragma unroll
    for (int off = 1; off < 32; off <<= 1) {
      int t = __shfl_up(pre, off, 32);
      if (lane >= off) pre += t;
    }
    int excl = pre - tot;
    int base = cnt + excl;
    int seen = 0;
#pragma unroll
    for (int e = 0; e < 4; ++e) {
      if (m4 & (1 << e)) {
        int pos = base + seen;
        if (pos < MAX_NBR) nbr_idx[row * MAX_NBR + pos] = c0 + 4 * lane + e;
        ++seen;
      }
    }
    int wavetot = tot;
#pragma unroll
    for (int off = 16; off > 0; off >>= 1) wavetot += __shfl_xor(wavetot, off, 32);
    cnt += wavetot;
  }
  if (lane == 0) nbr_cnt[row] = cnt < MAX_NBR ? cnt : MAX_NBR;
}

// ---------------------------------------------------------------------------
// WMMA fragment loads from LDS, per CDNA5 ISA 7.12.2 (wave32):
//  A 16x32 bf16: lane m=l&15, K-half=(l>>4)*8; elem j<8 -> K=kh+j, j>=8 -> kh+16+j-8
//  B 32x16 bf16: lane n=l&15, K-half=(l>>4)*8; elem j<8 -> K=kh+j, j>=8 -> kh+16+j-8
//  C/D f32:      lane n=l&15, VGPR e -> M = e + (l>>4)*8
// ---------------------------------------------------------------------------
__device__ __forceinline__ v16bf load_a_frag(const bf16_t* lA, int lane, int ldk) {
  v16bf a;
  int m = lane & 15;
  int kh = (lane >> 4) * 8;
  const bf16_t* p = lA + m * ldk + kh;
#pragma unroll
  for (int j = 0; j < 8; ++j) { a[j] = p[j]; a[j + 8] = p[16 + j]; }
  return a;
}
__device__ __forceinline__ v16bf load_b_frag(const bf16_t* lB, int lane, int ldn, int nofs) {
  v16bf b;
  const bf16_t* p = lB + ((lane >> 4) * 8) * ldn + nofs + (lane & 15);
#pragma unroll
  for (int j = 0; j < 8; ++j) { b[j] = p[j * ldn]; b[j + 8] = p[(16 + j) * ldn]; }
  return b;
}

// ---------------------------------------------------------------------------
// Encoder GEMM: h[8192][64] = xb[8192][3072]bf16 @ WencT + bias.
// 4 waves/block (one 16-col N-tile each); double-buffered async LDS staging,
// K chunk = 128 -> 4 WMMA steps per chunk, 24 chunks.
// ---------------------------------------------------------------------------
__device__ __forceinline__ void stage_enc(const bf16_t* __restrict__ xb,
                                          const bf16_t* __restrict__ wT,
                                          bf16_t* lA, bf16_t* lB,
                                          int row0, int kbase, int tid) {
#pragma unroll
  for (int i = tid; i < 256; i += 128) {       // A: 16 rows x 256B
    int m = i >> 4, c = (i & 15) * 8;
    async_copy16(xb + (size_t)(row0 + m) * ENC_KPAD + kbase + c, lA + m * ENC_KCH + c);
  }
#pragma unroll
  for (int i = tid; i < 1024; i += 128) {      // B: 128 rows x 128B
    int kr = i >> 3, c = (i & 7) * 8;
    async_copy16(wT + (size_t)(kbase + kr) * D_PAD + c, lB + kr * D_PAD + c);
  }
}

__global__ __launch_bounds__(128) void gemm_encoder(const bf16_t* __restrict__ xb,
                                                    const bf16_t* __restrict__ wT,
                                                    const float* __restrict__ biasPad,
                                                    float* __restrict__ hbuf) {
  __shared__ bf16_t lA[2][16 * ENC_KCH];     // 2 x 4 KB
  __shared__ bf16_t lB[2][ENC_KCH * D_PAD];  // 2 x 16 KB
  const int tid = threadIdx.x;
  const int wave = tid >> 5, lane = tid & 31;
  const int row0 = blockIdx.x * 16;
  v8f acc = {};
  stage_enc(xb, wT, lA[0], lB[0], row0, 0, tid);
  for (int kt = 0; kt < ENC_KCHUNKS; ++kt) {
    wait_async_lds();
    __syncthreads();
    if (kt + 1 < ENC_KCHUNKS)
      stage_enc(xb, wT, lA[(kt + 1) & 1], lB[(kt + 1) & 1], row0, (kt + 1) * ENC_KCH, tid);
    const bf16_t* cA = lA[kt & 1];
    const bf16_t* cB = lB[kt & 1];
#pragma unroll
    for (int ks = 0; ks < 4; ++ks) {
      v16bf a = load_a_frag(cA + ks * KCH, lane, ENC_KCH);
      v16bf b = load_b_frag(cB + ks * KCH * D_PAD, lane, D_PAD, wave * 16);
      acc = __builtin_amdgcn_wmma_f32_16x16x32_bf16(false, a, false, b, (short)0, acc,
                                                    false, false);
    }
  }
  int n = lane & 15, mb = (lane >> 4) * 8;
  int col = wave * 16 + n;
  float bia = biasPad[col];
#pragma unroll
  for (int e = 0; e < 8; ++e)
    hbuf[(size_t)(row0 + mb + e) * D_PAD + col] = acc[e] + bia;
}

// ---------------------------------------------------------------------------
// Decoder GEMM: out[8192][3000] = db(bf16)[8192][64] @ WdecT[64][3072].
// 8 waves/block = 128 cols; K=64 in two WMMA steps; async staged tiles.
// ---------------------------------------------------------------------------
__global__ __launch_bounds__(256) void gemm_decoder(const bf16_t* __restrict__ db,
                                                    const bf16_t* __restrict__ wdecT,
                                                    float* __restrict__ out) {
  __shared__ bf16_t lA[16 * D_PAD];   // 2 KB
  __shared__ bf16_t lB[D_PAD * 128];  // 16 KB
  const int tid = threadIdx.x;
  const int wave = tid >> 5, lane = tid & 31;
  const int row0 = blockIdx.x * 16;
  const int col0 = blockIdx.y * 128;
  for (int i = tid; i < 128; i += 256) {       // A: 16 rows x 128B
    int m = i >> 3, c = (i & 7) * 8;
    async_copy16(db + (size_t)(row0 + m) * D_PAD + c, lA + m * D_PAD + c);
  }
#pragma unroll
  for (int i = tid; i < 1024; i += 256) {      // B: 64 rows x 256B
    int kr = i >> 4, c = (i & 15) * 8;
    async_copy16(wdecT + (size_t)kr * DEC_OSTRIDE + col0 + c, lB + kr * 128 + c);
  }
  wait_async_lds();
  __syncthreads();
  v8f acc = {};
#pragma unroll
  for (int kc = 0; kc < 2; ++kc) {
    v16bf a = load_a_frag(lA + kc * KCH, lane, D_PAD);
    v16bf b = load_b_frag(lB + (size_t)kc * KCH * 128, lane, 128, wave * 16);
    acc = __builtin_amdgcn_wmma_f32_16x16x32_bf16(false, a, false, b, (short)0, acc,
                                                  false, false);
  }
  int n = lane & 15, mb = (lane >> 4) * 8;
  int col = col0 + wave * 16 + n;
  if (col < D_IN) {
#pragma unroll
    for (int e = 0; e < 8; ++e)
      out[(size_t)(row0 + mb + e) * D_IN + col] = acc[e];
  }
}

// ---------------------------------------------------------------------------
// q/k/v projection: 50x50 weights, 4 rows per 256-thread block. Pads zeroed.
// ---------------------------------------------------------------------------
__global__ __launch_bounds__(256) void proj_qkv(const float* __restrict__ xin,
    const float* __restrict__ wq, const float* __restrict__ wk,
    const float* __restrict__ wv, float* __restrict__ qb,
    float* __restrict__ kb, float* __restrict__ vb) {
  __shared__ float xs[4][D_LAT];
  int sub = threadIdx.x >> 6;
  int o = threadIdx.x & 63;
  int row = blockIdx.x * 4 + sub;
  if (o < D_LAT) xs[sub][o] = xin[(size_t)row * D_PAD + o];
  __syncthreads();
  float sq = 0.f, sk = 0.f, sv = 0.f;
  if (o < D_LAT) {
#pragma unroll
    for (int k2 = 0; k2 < D_LAT; ++k2) {
      float xv = xs[sub][k2];
      sq += xv * wq[o * D_LAT + k2];
      sk += xv * wk[o * D_LAT + k2];
      sv += xv * wv[o * D_LAT + k2];
    }
  }
  size_t oidx = (size_t)row * D_PAD + o;
  qb[oidx] = (o < D_LAT) ? sq : 0.f;
  kb[oidx] = (o < D_LAT) ? sk : 0.f;
  vb[oidx] = (o < D_LAT) ? sv : 0.f;
}

// ---------------------------------------------------------------------------
// Sparse masked attention: one wave32 per row; lane owns dims (2l, 2l+1).
// Matches dense NEG=-1e9 softmax exactly (exp underflow for masked entries).
// ---------------------------------------------------------------------------
__global__ __launch_bounds__(256) void sparse_attn(const float* __restrict__ qb,
    const float* __restrict__ kb, const float* __restrict__ vb,
    const int* __restrict__ nbr_idx, const int* __restrict__ nbr_cnt,
    float* __restrict__ outb) {
  int wave = threadIdx.x >> 5, lane = threadIdx.x & 31;
  int row = blockIdx.x * 8 + wave;
  int cnt = nbr_cnt[row];
  int d0 = 2 * lane;
  float q0 = qb[(size_t)row * D_PAD + d0];
  float q1 = qb[(size_t)row * D_PAD + d0 + 1];
  float sc[MAX_NBR];
#pragma unroll
  for (int c = 0; c < MAX_NBR; ++c) {
    float p = 0.f;
    if (c < cnt) {
      int j = nbr_idx[row * MAX_NBR + c];
      p = q0 * kb[(size_t)j * D_PAD + d0] + q1 * kb[(size_t)j * D_PAD + d0 + 1];
    }
#pragma unroll
    for (int off = 16; off > 0; off >>= 1) p += __shfl_xor(p, off, 32);
    sc[c] = (c < cnt) ? p * 0.14142135623730951f : -1e30f;  // 1/sqrt(50)
  }
  float mx = -1e30f;
#pragma unroll
  for (int c = 0; c < MAX_NBR; ++c) mx = fmaxf(mx, sc[c]);
  float w[MAX_NBR];
  float wsum = 0.f;
#pragma unroll
  for (int c = 0; c < MAX_NBR; ++c) {
    float e = __expf(sc[c] - mx);
    w[c] = e;
    wsum += e;
  }
  float inv = 1.f / wsum;
  float o0 = 0.f, o1 = 0.f;
#pragma unroll
  for (int c = 0; c < MAX_NBR; ++c) {
    if (c < cnt) {
      int j = nbr_idx[row * MAX_NBR + c];
      float wc = w[c] * inv;
      o0 += wc * vb[(size_t)j * D_PAD + d0];
      o1 += wc * vb[(size_t)j * D_PAD + d0 + 1];
    }
  }
  outb[(size_t)row * D_PAD + d0] = o0;
  outb[(size_t)row * D_PAD + d0 + 1] = o1;
}

// ---------------------------------------------------------------------------
// mu / logvar / reparameterized z (z padded to 64 cols with zeros)
// ---------------------------------------------------------------------------
__global__ __launch_bounds__(256) void latent_kernel(const float* __restrict__ h,
    const float* __restrict__ mu_w, const float* __restrict__ mu_b,
    const float* __restrict__ lv_w, const float* __restrict__ lv_b,
    const float* __restrict__ eps, float* __restrict__ mu_out,
    float* __restrict__ lv_out, float* __restrict__ zb) {
  __shared__ float xs[4][D_LAT];
  int sub = threadIdx.x >> 6;
  int o = threadIdx.x & 63;
  int row = blockIdx.x * 4 + sub;
  if (o < D_LAT) xs[sub][o] = h[(size_t)row * D_PAD + o];
  __syncthreads();
  if (o < D_LAT) {
    float sm = mu_b[o], sl = lv_b[o];
#pragma unroll
    for (int k2 = 0; k2 < D_LAT; ++k2) {
      float xv = xs[sub][k2];
      sm += xv * mu_w[o * D_LAT + k2];
      sl += xv * lv_w[o * D_LAT + k2];
    }
    mu_out[(size_t)row * D_LAT + o] = sm;
    lv_out[(size_t)row * D_LAT + o] = sl;
    zb[(size_t)row * D_PAD + o] = sm + eps[(size_t)row * D_LAT + o] * __expf(0.5f * sl);
  } else {
    zb[(size_t)row * D_PAD + o] = 0.f;
  }
}

// ---------------------------------------------------------------------------
extern "C" void kernel_launch(void* const* d_in, const int* in_sizes, int n_in,
                              void* d_out, int out_size, void* d_ws, size_t ws_size,
                              hipStream_t stream) {
  const float* x    = (const float*)d_in[0];
  const float* knn  = (const float*)d_in[1];
  const float* eps  = (const float*)d_in[2];
  const float* fc1w = (const float*)d_in[3];
  const float* fc1b = (const float*)d_in[4];
  const float* fc2w = (const float*)d_in[5];
  const float* mu_w = (const float*)d_in[6];
  const float* mu_b = (const float*)d_in[7];
  const float* lv_w = (const float*)d_in[8];
  const float* lv_b = (const float*)d_in[9];
  const float* decw = (const float*)d_in[10];
  const float* e1q = (const float*)d_in[11];
  const float* e1k = (const float*)d_in[12];
  const float* e1v = (const float*)d_in[13];
  const float* e2q = (const float*)d_in[14];
  const float* e2k = (const float*)d_in[15];
  const float* e2v = (const float*)d_in[16];
  const float* d1q = (const float*)d_in[17];
  const float* d1k = (const float*)d_in[18];
  const float* d1v = (const float*)d_in[19];
  const float* d2q = (const float*)d_in[20];
  const float* d2k = (const float*)d_in[21];
  const float* d2v = (const float*)d_in[22];

  char* wsp = (char*)d_ws;
  auto alloc = [&](size_t bytes) -> char* {
    char* p = wsp;
    wsp += (bytes + 255) & ~(size_t)255;
    return p;
  };
  bf16_t* wencT = (bf16_t*)alloc((size_t)ENC_KPAD * D_PAD * sizeof(bf16_t));
  float*  biasP = (float*)alloc(D_PAD * sizeof(float));
  bf16_t* wdecT = (bf16_t*)alloc((size_t)D_PAD * DEC_OSTRIDE * sizeof(bf16_t));
  bf16_t* xb    = (bf16_t*)alloc((size_t)N_SPOTS * ENC_KPAD * sizeof(bf16_t));
  bf16_t* dbufb = (bf16_t*)alloc((size_t)N_SPOTS * D_PAD * sizeof(bf16_t));
  int*    ncnt  = (int*)alloc(N_SPOTS * sizeof(int));
  int*    nidx  = (int*)alloc((size_t)N_SPOTS * MAX_NBR * sizeof(int));
  float*  hbuf  = (float*)alloc((size_t)N_SPOTS * D_PAD * sizeof(float));
  float*  qb    = (float*)alloc((size_t)N_SPOTS * D_PAD * sizeof(float));
  float*  kb    = (float*)alloc((size_t)N_SPOTS * D_PAD * sizeof(float));
  float*  vb    = (float*)alloc((size_t)N_SPOTS * D_PAD * sizeof(float));
  float*  buf1  = (float*)alloc((size_t)N_SPOTS * D_PAD * sizeof(float));
  float*  buf2  = (float*)alloc((size_t)N_SPOTS * D_PAD * sizeof(float));
  float*  zbuf  = (float*)alloc((size_t)N_SPOTS * D_PAD * sizeof(float));
  (void)ws_size; (void)in_sizes; (void)n_in; (void)out_size;

  float* out_x  = (float*)d_out;
  float* out_mu = out_x + (size_t)N_SPOTS * D_IN;
  float* out_lv = out_mu + (size_t)N_SPOTS * D_LAT;

  prep_weights<<<512, 256, 0, stream>>>(fc1w, fc1b, fc2w, decw, wencT, biasP, wdecT);
  convert_pad_bf16<<<2048, 256, 0, stream>>>(x, xb, N_SPOTS, D_IN, ENC_KPAD);
  build_nbrs<<<N_SPOTS / 8, 256, 0, stream>>>(knn, nidx, ncnt);
  gemm_encoder<<<N_SPOTS / 16, 128, 0, stream>>>(xb, wencT, biasP, hbuf);

  proj_qkv<<<N_SPOTS / 4, 256, 0, stream>>>(hbuf, e1q, e1k, e1v, qb, kb, vb);
  sparse_attn<<<N_SPOTS / 8, 256, 0, stream>>>(qb, kb, vb, nidx, ncnt, buf1);
  proj_qkv<<<N_SPOTS / 4, 256, 0, stream>>>(buf1, e2q, e2k, e2v, qb, kb, vb);
  sparse_attn<<<N_SPOTS / 8, 256, 0, stream>>>(qb, kb, vb, nidx, ncnt, buf2);

  latent_kernel<<<N_SPOTS / 4, 256, 0, stream>>>(buf2, mu_w, mu_b, lv_w, lv_b, eps,
                                                 out_mu, out_lv, zbuf);

  proj_qkv<<<N_SPOTS / 4, 256, 0, stream>>>(zbuf, d1q, d1k, d1v, qb, kb, vb);
  sparse_attn<<<N_SPOTS / 8, 256, 0, stream>>>(qb, kb, vb, nidx, ncnt, buf1);
  proj_qkv<<<N_SPOTS / 4, 256, 0, stream>>>(buf1, d2q, d2k, d2v, qb, kb, vb);
  sparse_attn<<<N_SPOTS / 8, 256, 0, stream>>>(qb, kb, vb, nidx, ncnt, buf2);

  convert_pad_bf16<<<256, 256, 0, stream>>>(buf2, dbufb, N_SPOTS, D_PAD, D_PAD);
  gemm_decoder<<<dim3(N_SPOTS / 16, (D_IN + 127) / 128), 256, 0, stream>>>(dbufb, wdecT,
                                                                           out_x);
}
